// LightGCNAttention_62405874811874
// MI455X (gfx1250) — matrix operand order repaired and checked by reading, not compiled, not run on hardware
//
#include <hip/hip_runtime.h>
#include <stdint.h>

#define N_USERS 120000
#define N_ITEMS 80000
#define N_NODES (N_USERS + N_ITEMS)
#define DIM 64
#define N_EDGES 1280000
#define NLAYERS 3
#define BATCH 4096
#define NODES_PER_BLOCK 64
#define ATTN_BLOCKS (N_NODES / NODES_PER_BLOCK)   // 3125, exact
static_assert(N_NODES % NODES_PER_BLOCK == 0, "exact grid");
static_assert(N_USERS % 16 == 0, "U/V boundary tile aligned");

typedef __bf16 v16bf __attribute__((ext_vector_type(16)));
typedef float  v8f   __attribute__((ext_vector_type(8)));
typedef unsigned int u32x4 __attribute__((ext_vector_type(4)));
typedef int i32x4 __attribute__((ext_vector_type(4)));
typedef int i32x8 __attribute__((ext_vector_type(8)));

#if defined(__gfx1250__) && __has_builtin(__builtin_amdgcn_tensor_load_to_lds)
#define HAVE_TDM 1
#else
#define HAVE_TDM 0
#endif

// ---------------- WMMA fragment helpers (bf16 16x16x32, wave32) ----------------

// A: 16x32 tile from fp32 source (ld = DIM). lane L: row m=L&15, kb=(L>=16)?8:0;
// elems 0..7 -> K=kb+e ; elems 8..15 -> K=kb+16+(e-8). Two 32B runs -> b128 loads.
__device__ __forceinline__ v16bf load_a_frag(const float* base, int koff, int lane) {
  const int m  = lane & 15;
  const int kb = (lane >> 4) * 8;
  const float* rp = base + m * DIM + koff + kb;
  float4 x0 = *(const float4*)(rp);
  float4 x1 = *(const float4*)(rp + 4);
  float4 x2 = *(const float4*)(rp + 16);
  float4 x3 = *(const float4*)(rp + 20);
  v16bf a;
  a[0]  = (__bf16)x0.x; a[1]  = (__bf16)x0.y; a[2]  = (__bf16)x0.z; a[3]  = (__bf16)x0.w;
  a[4]  = (__bf16)x1.x; a[5]  = (__bf16)x1.y; a[6]  = (__bf16)x1.z; a[7]  = (__bf16)x1.w;
  a[8]  = (__bf16)x2.x; a[9]  = (__bf16)x2.y; a[10] = (__bf16)x2.z; a[11] = (__bf16)x2.w;
  a[12] = (__bf16)x3.x; a[13] = (__bf16)x3.y; a[14] = (__bf16)x3.z; a[15] = (__bf16)x3.w;
  return a;
}

// D = A(16x64) x W(64x64): K split 2x32, N split 4x16. B frags are pre-packed bf16
// (per-lane contiguous 32B) -> two b128 loads each. Store fp32 D tile (ld = DIM).
__device__ __forceinline__ void gemm16x64_store(v16bf a0, v16bf a1,
                                                const __bf16* __restrict__ wp,
                                                float* dst, int lane) {
  const int cm = (lane >> 4) * 8;
  const int cn = lane & 15;
#pragma unroll
  for (int nt = 0; nt < 4; ++nt) {
    v16bf b0 = *(const v16bf*)(wp + ((0 * 4 + nt) * 32 + lane) * 16);
    v16bf b1 = *(const v16bf*)(wp + ((1 * 4 + nt) * 32 + lane) * 16);
    v8f acc = {};
    acc = __builtin_amdgcn_wmma_f32_16x16x32_bf16(false, a0, false, b0,
                                                  (short)0, acc, false, false);
    acc = __builtin_amdgcn_wmma_f32_16x16x32_bf16(false, a1, false, b1,
                                                  (short)0, acc, false, false);
#pragma unroll
    for (int r = 0; r < 8; ++r) dst[(cm + r) * DIM + nt * 16 + cn] = acc[r];
  }
}

// ---------------------------------- kernels ----------------------------------

// Pre-convert the four 64x64 fp32 weights into per-lane bf16 B-fragment layout:
// Wpk[w][kc][nt][lane][16]  (w: q,k,v,o). One thread per (w,kc,nt,lane).
__global__ void pack_weights_kernel(const float* __restrict__ Wq, const float* __restrict__ Wk,
                                    const float* __restrict__ Wv, const float* __restrict__ Wo,
                                    __bf16* __restrict__ Wpk) {
  int t = blockIdx.x * blockDim.x + threadIdx.x;
  if (t >= 4 * 2 * 4 * 32) return;
  int lane = t & 31;
  int nt   = (t >> 5) & 3;
  int kc   = (t >> 7) & 1;
  int w    = (t >> 8) & 3;
  const float* W = (w == 0) ? Wq : (w == 1) ? Wk : (w == 2) ? Wv : Wo;
  const int n  = lane & 15;
  const int kb = (lane >> 4) * 8;
  const float* cp = W + (kc * 32 + kb) * DIM + nt * 16 + n;
  __bf16* dst = Wpk + (size_t)t * 16;
#pragma unroll
  for (int e = 0; e < 16; ++e) dst[e] = (__bf16)cp[(e < 8 ? e : e + 8) * DIM];
}

__global__ void concat_kernel(const float* __restrict__ U, const float* __restrict__ V,
                              float* __restrict__ E0) {
  int t = blockIdx.x * blockDim.x + threadIdx.x;
  const int total = N_NODES * DIM / 4;
  if (t >= total) return;
  int off = t * 4;
  float4 v;
  if (off < N_USERS * DIM) v = *(const float4*)(U + off);
  else                     v = *(const float4*)(V + (off - N_USERS * DIM));
  *(float4*)(E0 + off) = v;
}

__global__ void zero_kernel(float* __restrict__ p, int n4) {
  int t = blockIdx.x * blockDim.x + threadIdx.x;
  if (t < n4) *(float4*)(p + t * 4) = make_float4(0.f, 0.f, 0.f, 0.f);
}

// One float4 feature-chunk of one edge per thread (16 threads/edge).
// Whole 51.2MB layer fits in 192MB L2 -> atomics stay L2-resident.
__global__ void prop_kernel(const int* __restrict__ esrc, const int* __restrict__ edst,
                            const float* __restrict__ eval, const float* __restrict__ Ein,
                            float* __restrict__ Eout) {
  long t = (long)blockIdx.x * blockDim.x + threadIdx.x;
  if (t >= (long)N_EDGES * (DIM / 4)) return;
  int e = (int)(t >> 4);
  int c = ((int)t & 15) * 4;
  int s = esrc[e];
  int d = edst[e];
  float v = eval[e];
  float4 x = *(const float4*)(Ein + (long)s * DIM + c);
  float* o = Eout + (long)d * DIM + c;
  atomicAdd(o + 0, v * x.x);
  atomicAdd(o + 1, v * x.y);
  atomicAdd(o + 2, v * x.z);
  atomicAdd(o + 3, v * x.w);
}

__global__ __launch_bounds__(128) void attn_kernel(const float* __restrict__ Es,
                                                   const __bf16* __restrict__ Wpk,
                                                   float* __restrict__ Eout) {
  __shared__ float etile[4 * NODES_PER_BLOCK * DIM];  // 64KB: 4 layers x 64 nodes x 64
  __shared__ float scratch[4 * 2048];                 // 32KB: per-wave Q/heads + K/V tile
  const int wave = threadIdx.x >> 5;
  const int lane = threadIdx.x & 31;
  const int nodeBase = blockIdx.x * NODES_PER_BLOCK;

  // ---- stage the block's 64x64x4-layer tile into LDS ----
#if HAVE_TDM
  if (wave == 0) {
    // D# describing a 3D tile: x=64 feats (contig), y=64 nodes (stride 64),
    // z=4 layers (stride N_NODES*64), fp32 elements, destination = etile.
    unsigned long long ga = (unsigned long long)(uintptr_t)(Es + (size_t)nodeBase * DIM);
    unsigned ldsoff = (unsigned)(uintptr_t)(__attribute__((address_space(3))) float*)etile;
    unsigned long long zs = (unsigned long long)N_NODES * DIM;  // layer stride (elems)
    u32x4 g0;
    g0[0] = 1u;                                            // count=1, gather off
    g0[1] = ldsoff;                                        // lds_addr (bytes)
    g0[2] = (unsigned)ga;                                  // global_addr[31:0]
    g0[3] = (unsigned)((ga >> 32) & 0x01FFFFFFu) | (2u << 30);  // addr[56:32] | type=2
    i32x8 g1;
    g1[0] = (int)(2u << 16);                               // data_size = 4B
    g1[1] = (int)(64u << 16);                              // tensor_dim0 = 64
    g1[2] = (int)(64u << 16);                              // tensor_dim1 = 64
    g1[3] = (int)(64u << 16);                              // tile_dim0 = 64
    g1[4] = (int)(64u | (4u << 16));                       // tile_dim1=64, tile_dim2=4
    g1[5] = 64;                                            // tensor_dim0_stride lo32
    g1[6] = (int)(((unsigned)(zs & 0xFFFFu)) << 16);       // dim0_str hi16=0 | dim1_str[15:0]
    g1[7] = (int)(zs >> 16);                               // dim1_stride[47:16]
    i32x4 g2; g2[0] = 4; g2[1] = 0; g2[2] = 0; g2[3] = 0;  // tensor_dim2 = 4
    i32x4 g3; g3[0] = 0; g3[1] = 0; g3[2] = 0; g3[3] = 0;
#if __clang_major__ >= 23
    i32x8 g4 = {};
    __builtin_amdgcn_tensor_load_to_lds(g0, g1, g2, g3, g4, 0);
#else
    __builtin_amdgcn_tensor_load_to_lds(g0, g1, g2, g3, 0);
#endif
#if __has_builtin(__builtin_amdgcn_s_wait_tensorcnt)
    __builtin_amdgcn_s_wait_tensorcnt(0);
#else
    asm volatile("s_wait_tensorcnt 0x0" ::: "memory");
#endif
  }
  __syncthreads();
#else
  // fallback: cooperative coalesced float4 staging (tile rows are contiguous)
#pragma unroll
  for (int l = 0; l < 4; ++l) {
    const float4* srcl = (const float4*)(Es + ((size_t)l * N_NODES + nodeBase) * DIM);
    float4* dstl = (float4*)(etile + l * NODES_PER_BLOCK * DIM);
    for (int j = threadIdx.x; j < NODES_PER_BLOCK * DIM / 4; j += 128) dstl[j] = srcl[j];
  }
  __syncthreads();
#endif

  // A fragments for all 4 layers, held in registers (layer 0 doubles as Q input)
  v16bf a0[4], a1[4];
#pragma unroll
  for (int l = 0; l < 4; ++l) {
    const float* bt = etile + l * NODES_PER_BLOCK * DIM + wave * 16 * DIM;
    a0[l] = load_a_frag(bt, 0, lane);
    a1[l] = load_a_frag(bt, 32, lane);
  }

  float* qbuf  = scratch + wave * 2048;   // 16x64 Q tile, later reused for heads
  float* kvbuf = qbuf + 1024;             // 16x64 K_l / V_l / out scratch
  const __bf16* WpkQ = Wpk;               // each packed weight = 2*4*32*16 = 4096 bf16
  const __bf16* WpkK = Wpk + 4096;
  const __bf16* WpkV = Wpk + 8192;
  const __bf16* WpkO = Wpk + 12288;

  gemm16x64_store(a0[0], a1[0], WpkQ, qbuf, lane);

  const int node = lane & 15;             // lane <-> (node, head)
  const int head = lane >> 4;
  const float4* q4 = (const float4*)(qbuf + node * DIM + head * 32);
  const float4* kv4 = (const float4*)(kvbuf + node * DIM + head * 32);

  // scores over 4 layers (K pass; per-wave LDS region, in-order DS => no barriers)
  float sc[4];
#pragma unroll
  for (int l = 0; l < 4; ++l) {
    gemm16x64_store(a0[l], a1[l], WpkK, kvbuf, lane);
    float s = 0.f;
#pragma unroll
    for (int j = 0; j < 8; ++j) {
      float4 qa = q4[j], ka = kv4[j];
      s += qa.x * ka.x + qa.y * ka.y + qa.z * ka.z + qa.w * ka.w;
    }
    sc[l] = s * 0.17677669529663687f;     // 1/sqrt(32), TEMPERATURE=1
  }

  // softmax over layers
  float mx = fmaxf(fmaxf(sc[0], sc[1]), fmaxf(sc[2], sc[3]));
  float al[4], ssum = 0.f;
#pragma unroll
  for (int l = 0; l < 4; ++l) { al[l] = __expf(sc[l] - mx); ssum += al[l]; }
  const float inv = 1.f / ssum;

  // heads accumulation (V pass)
  float hv[32];
#pragma unroll
  for (int v = 0; v < 32; ++v) hv[v] = 0.f;
#pragma unroll
  for (int l = 0; l < 4; ++l) {
    gemm16x64_store(a0[l], a1[l], WpkV, kvbuf, lane);
    const float w = al[l] * inv;
#pragma unroll
    for (int j = 0; j < 8; ++j) {
      float4 va = kv4[j];
      hv[4 * j + 0] += w * va.x;
      hv[4 * j + 1] += w * va.y;
      hv[4 * j + 2] += w * va.z;
      hv[4 * j + 3] += w * va.w;
    }
  }

  // heads -> qbuf (reuse), output projection, coalesced b128 store via LDS bounce
#pragma unroll
  for (int v = 0; v < 32; ++v) qbuf[node * DIM + head * 32 + v] = hv[v];

  v16bf h0 = load_a_frag(qbuf, 0, lane);
  v16bf h1 = load_a_frag(qbuf, 32, lane);
  gemm16x64_store(h0, h1, WpkO, kvbuf, lane);

  const float4* s4 = (const float4*)kvbuf;
  float4* g4 = (float4*)(Eout + (size_t)(nodeBase + wave * 16) * DIM);
#pragma unroll
  for (int j = 0; j < 8; ++j) g4[j * 32 + lane] = s4[j * 32 + lane];
}

// one wave per (u,i) pair, float2 per lane, shuffle reduce
__global__ void score_kernel(const float* __restrict__ Eout, const int* __restrict__ u,
                             const int* __restrict__ ii, float* __restrict__ out) {
  int gid = blockIdx.x * blockDim.x + threadIdx.x;
  int w = gid >> 5;
  int lane = gid & 31;
  if (w >= BATCH) return;
  const float2* a = (const float2*)(Eout + (long)u[w] * DIM);
  const float2* b = (const float2*)(Eout + ((long)N_USERS + ii[w]) * DIM);
  float2 x = a[lane];
  float2 y = b[lane];
  float s = x.x * y.x + x.y * y.y;
#pragma unroll
  for (int off = 16; off > 0; off >>= 1) s += __shfl_down(s, off, 32);
  if (lane == 0) out[w] = s;
}

// --------------------------------- launcher ---------------------------------

extern "C" void kernel_launch(void* const* d_in, const int* in_sizes, int n_in,
                              void* d_out, int out_size, void* d_ws, size_t ws_size,
                              hipStream_t stream) {
  const int*   edge_src = (const int*)d_in[0];
  const int*   edge_dst = (const int*)d_in[1];
  const float* edge_val = (const float*)d_in[2];
  const float* U_emb    = (const float*)d_in[3];
  const float* V_emb    = (const float*)d_in[4];
  const float* Wq       = (const float*)d_in[5];
  const float* Wk       = (const float*)d_in[6];
  const float* Wv       = (const float*)d_in[7];
  const float* Wo       = (const float*)d_in[8];
  const int*   u        = (const int*)d_in[9];
  const int*   ii       = (const int*)d_in[10];
  float*       out      = (float*)d_out;

  // ws: Es[4][N][D] fp32 (204.8MB) + E_out[N][D] fp32 (51.2MB) + packed weights (32KB)
  float*  Es   = (float*)d_ws;
  float*  Eout = Es + (size_t)4 * N_NODES * DIM;
  __bf16* Wpk  = (__bf16*)(Eout + (size_t)N_NODES * DIM);

  pack_weights_kernel<<<4, 256, 0, stream>>>(Wq, Wk, Wv, Wo, Wpk);

  const int ND4 = N_NODES * DIM / 4;
  concat_kernel<<<(ND4 + 255) / 256, 256, 0, stream>>>(U_emb, V_emb, Es);

  for (int l = 1; l <= NLAYERS; ++l) {
    float* Ein = Es + (size_t)(l - 1) * N_NODES * DIM;
    float* Enx = Es + (size_t)l * N_NODES * DIM;
    zero_kernel<<<(ND4 + 255) / 256, 256, 0, stream>>>(Enx, ND4);
    long pt = (long)N_EDGES * (DIM / 4);
    prop_kernel<<<(int)((pt + 255) / 256), 256, 0, stream>>>(edge_src, edge_dst, edge_val, Ein, Enx);
  }

  attn_kernel<<<ATTN_BLOCKS, 128, 0, stream>>>(Es, Wpk, Eout);
  score_kernel<<<(BATCH * 32 + 255) / 256, 256, 0, stream>>>(Eout, u, ii, out);

  (void)in_sizes; (void)n_in; (void)out_size; (void)ws_size;
}
